// model_rel_gru_57861799411874
// MI455X (gfx1250) — matrix operand order repaired and checked by reading, not compiled, not run on hardware
//
#include <hip/hip_runtime.h>
#include <hip/hip_bf16.h>
#include <math.h>

typedef __attribute__((ext_vector_type(16))) __bf16 v16bf;
typedef __attribute__((ext_vector_type(8)))  float  v8f;

// Model dims (fixed by the reference)
// B=64 scenes, APER=16, A=1024 agents, C=256, M=128, N=256, H=5, D=384
// GEMM N-dims padded to multiples of 64 columns (4 WMMA tiles) so the inner
// WMMA loop is branch-free: write head 385->448, read head 645->704.

// ---------------------------------------------------------------------------
// elementwise / setup kernels
// ---------------------------------------------------------------------------

__global__ void zero_kernel(float* p, long n) {
  long i = (long)blockIdx.x * blockDim.x + threadIdx.x;
  if (i < n) p[i] = 0.0f;
}

__global__ void bcast_kernel(float* dst, const float* __restrict__ src, long rows, long cols) {
  long i = (long)blockIdx.x * blockDim.x + threadIdx.x;
  if (i < rows * cols) dst[i] = src[i % cols];
}

// 2 -> relu(8) -> 16 MLP over (A, 20, 2) trajectories; out layout (T, A, 16)
__global__ void embed_kernel(const float* __restrict__ traj,
                             const float* __restrict__ w1, const float* __restrict__ b1,
                             const float* __restrict__ w2, const float* __restrict__ b2,
                             float* __restrict__ out) {
  int idx = blockIdx.x * blockDim.x + threadIdx.x;
  if (idx >= 1024 * 20) return;
  int g = idx % 1024, t = idx / 1024;
  float x0 = traj[((long)g * 20 + t) * 2 + 0];
  float x1 = traj[((long)g * 20 + t) * 2 + 1];
  float hid[8];
#pragma unroll
  for (int i = 0; i < 8; ++i) {
    float v = w1[i * 2] * x0 + w1[i * 2 + 1] * x1 + b1[i];
    hid[i] = v > 0.0f ? v : 0.0f;
  }
  float* o = out + ((long)t * 1024 + g) * 16;
#pragma unroll
  for (int e = 0; e < 16; ++e) {
    float v = b2[e];
#pragma unroll
    for (int i = 0; i < 8; ++i) v += w2[e * 8 + i] * hid[i];
    o[e] = v;
  }
}

// Pack torch-linear weight W (Nout x K, row-major f32) into this file's B-fragment
// order (contiguous k, same as A) for v_wmma_f32_16x16x32_bf16 (B is K x Nout = W^T):
//   lane<16  : col = nt*16+lane, rows kt*32 + {2v, 2v+1}
//   lane>=16 : same col,        rows kt*32 + 16 + {2v, 2v+1}
// Zero-pads K to Kt*32 and Nout to Nt*16 (Nt is a multiple of 4).
__global__ void pack_b_kernel(const float* __restrict__ W, __bf16* __restrict__ P,
                              int K, int Nout, int Kt, int Nt) {
  int idx = blockIdx.x * blockDim.x + threadIdx.x;
  int total = Kt * Nt * 32;
  if (idx >= total) return;
  int lane = idx & 31;
  int tile = idx >> 5;
  int nt = tile % Nt;
  int kt = tile / Nt;
  int n = nt * 16 + (lane & 15);
  int kbase = kt * 32 + ((lane >> 4) ? 16 : 0);
  __bf16* dst = P + (long)idx * 16;
#pragma unroll
  for (int v = 0; v < 8; ++v) {
    int k0 = kbase + 2 * v;
    float f0 = (n < Nout && k0 < K) ? W[(long)n * K + k0] : 0.0f;
    float f1 = (n < Nout && k0 + 1 < K) ? W[(long)n * K + k0 + 1] : 0.0f;
    dst[2 * v] = (__bf16)f0;
    dst[2 * v + 1] = (__bf16)f1;
  }
}

// ---------------------------------------------------------------------------
// WMMA GEMM: C(M x Ntiles*16) = A(M x K, f32) * packed-B(bf16) + bias
// One wave computes a 16x64 output strip (A fragment reused across 4 B tiles).
// Ntiles % 4 == 0 (B zero-padded) -> branch-free WMMA chain.
// K loop split: K/32 full tiles (branch-free A loads) + one guarded tail tile
// (K % 8 == 0 holds for all call sites, so tail chunks are all-or-nothing).
// ---------------------------------------------------------------------------
__global__ void wmma_gemm_bias(const float* __restrict__ Amat, int lda, int Mrows,
                               const __bf16* __restrict__ Bp, int K, int Kt, int Ntiles,
                               const float* __restrict__ bias, int Nreal,
                               float* __restrict__ Cmat, int ldc) {
  const int wave = (int)(((long)blockIdx.x * blockDim.x + threadIdx.x) >> 5);
  const int lane = threadIdx.x & 31;
  const int Nt4 = Ntiles >> 2;
  const int Mt = Mrows >> 4;
  if (wave >= Mt * Nt4) return;
  const int mt = wave / Nt4;
  const int n4 = wave % Nt4;
  const int half = lane >> 4;
  const int l15 = lane & 15;
  const float* arow = Amat + (long)(mt * 16 + l15) * lda;

  v8f acc[4];
#pragma unroll
  for (int i = 0; i < 4; ++i)
#pragma unroll
    for (int r = 0; r < 8; ++r) acc[i][r] = 0.0f;

  const int KtFull = K >> 5;  // tiles with all 32 K-values in range

  for (int kt = 0; kt < KtFull; ++kt) {
    // A fragment: lane l15 = row, half selects K sub-range; f32 -> bf16 on the fly
    v16bf afrag;
#pragma unroll
    for (int j = 0; j < 2; ++j) {
      const int kb = kt * 32 + half * 16 + j * 8;
#pragma unroll
      for (int u = 0; u < 8; ++u) afrag[j * 8 + u] = (__bf16)arow[kb + u];
    }
    const __bf16* bbase = Bp + (((long)kt * Ntiles + n4 * 4) * 32 + lane) * 16;
    if (kt + 1 < Kt)  // pull next K-panel of B toward the WGP (global_prefetch_b8)
      __builtin_prefetch(Bp + (((long)(kt + 1) * Ntiles + n4 * 4) * 32 + lane) * 16, 0, 1);
    v16bf bf[4];
#pragma unroll
    for (int i = 0; i < 4; ++i) bf[i] = *(const v16bf*)(bbase + (long)i * 32 * 16);
#pragma unroll
    for (int i = 0; i < 4; ++i)
      acc[i] = __builtin_amdgcn_wmma_f32_16x16x32_bf16(false, afrag, false, bf[i],
                                                       (short)0, acc[i], false, false);
  }

  if (KtFull < Kt) {  // single zero-padded tail tile (K not a multiple of 32)
    const int kt = KtFull;
    v16bf afrag;
#pragma unroll
    for (int j = 0; j < 2; ++j) {
      const int kb = kt * 32 + half * 16 + j * 8;
      if (kb + 8 <= K) {
#pragma unroll
        for (int u = 0; u < 8; ++u) afrag[j * 8 + u] = (__bf16)arow[kb + u];
      } else {
#pragma unroll
        for (int u = 0; u < 8; ++u) afrag[j * 8 + u] = (__bf16)0.0f;
      }
    }
    const __bf16* bbase = Bp + (((long)kt * Ntiles + n4 * 4) * 32 + lane) * 16;
    v16bf bf[4];
#pragma unroll
    for (int i = 0; i < 4; ++i) bf[i] = *(const v16bf*)(bbase + (long)i * 32 * 16);
#pragma unroll
    for (int i = 0; i < 4; ++i)
      acc[i] = __builtin_amdgcn_wmma_f32_16x16x32_bf16(false, afrag, false, bf[i],
                                                       (short)0, acc[i], false, false);
  }

  // C layout: vgpr r -> row mt*16 + half*8 + r, col nt*16 + l15
#pragma unroll
  for (int i = 0; i < 4; ++i) {
    int col = (n4 * 4 + i) * 16 + l15;
    float bv = (col < Nreal) ? bias[col] : 0.0f;
    float* cbase = Cmat + (long)(mt * 16 + half * 8) * ldc + col;
#pragma unroll
    for (int r = 0; r < 8; ++r) cbase[(long)r * ldc] = acc[i][r] + bv;
  }
}

// ---------------------------------------------------------------------------
// GRU gate combine: h = (1-z)*n + z*h  with gi/gh holding [r|z|n] pre-acts.
// ---------------------------------------------------------------------------
__global__ void gru_combine_kernel(const float* __restrict__ gi, const float* __restrict__ gh,
                                   float* __restrict__ h, int rows, int S) {
  long idx = (long)blockIdx.x * blockDim.x + threadIdx.x;
  if (idx >= (long)rows * S) return;
  int row = (int)(idx / S), s = (int)(idx % S);
  const float* gir = gi + (long)row * 3 * S;
  const float* ghr = gh + (long)row * 3 * S;
  float r = 1.0f / (1.0f + __expf(-(gir[s] + ghr[s])));
  float z = 1.0f / (1.0f + __expf(-(gir[S + s] + ghr[S + s])));
  float n = tanhf(gir[2 * S + s] + r * ghr[2 * S + s]);
  h[idx] = (1.0f - z) * n + z * h[idx];
}

// xcat = [emb(16) | r(128)] per agent
__global__ void xcat_kernel(const float* __restrict__ emb, const float* __restrict__ rbuf,
                            float* __restrict__ xcat) {
  int idx = blockIdx.x * blockDim.x + threadIdx.x;
  if (idx >= 1024 * 144) return;
  int g = idx / 144, j = idx % 144;
  xcat[idx] = (j < 16) ? emb[g * 16 + j] : rbuf[g * 128 + (j - 16)];
}

// ---------------------------------------------------------------------------
// Write-head addressing. One block (128 thr) per agent. Each agent's write
// softmax is confined to its own 16-slot band (mask makes others exp(-1e9)=0).
// wp row layout: [0:128) k, [128] beta, [129:257) erase, [257:385) add, ld=448
// ---------------------------------------------------------------------------
__global__ void write_addr_kernel(const float* __restrict__ wp, const float* __restrict__ mem,
                                  float* __restrict__ ww) {
  __shared__ float kw[128];
  __shared__ float red[128];
  __shared__ float sims[16];
  int g = blockIdx.x;
  int t = threadIdx.x;
  int b = g >> 4, a = g & 15;
  const float* wrow = wp + (long)g * 448;
  float kv = tanhf(wrow[t]);
  kw[t] = kv;
  red[t] = kv * kv;
  __syncthreads();
  for (int s = 64; s > 0; s >>= 1) {
    if (t < s) red[t] += red[t + s];
    __syncthreads();
  }
  float knorm = sqrtf(red[0]) + 1e-16f;
  float bet = wrow[128];
  float beta = (bet > 20.0f) ? bet : log1pf(__expf(bet));  // softplus
  if (t < 16) {
    int n = a * 16 + t;
    const float* mrow = mem + ((long)b * 256 + n) * 128;
    float dot = 0.0f, nn = 0.0f;
    for (int m = 0; m < 128; ++m) {
      float mv = mrow[m];
      dot += kw[m] * mv;
      nn += mv * mv;
    }
    sims[t] = beta * dot / (knorm * (sqrtf(nn) + 1e-16f));
  }
  __syncthreads();
  if (t < 16) {
    float mx = sims[0];
#pragma unroll
    for (int i = 1; i < 16; ++i) mx = fmaxf(mx, sims[i]);
    float ssum = 0.0f;
#pragma unroll
    for (int i = 0; i < 16; ++i) ssum += __expf(sims[i] - mx);
    ww[g * 16 + t] = __expf(sims[t] - mx) / ssum;
  }
}

// mem[b,n,m] = mem*(1 - w*erase) + w*add ; bands are disjoint -> owner-only.
__global__ void mem_update_kernel(float* __restrict__ mem, const float* __restrict__ ww,
                                  const float* __restrict__ wp) {
  long idx = (long)blockIdx.x * blockDim.x + threadIdx.x;
  if (idx >= (long)64 * 256 * 128) return;
  int m = (int)(idx & 127);
  int n = (int)((idx >> 7) & 255);
  int b = (int)(idx >> 15);
  int a = n >> 4;
  int slot = n & 15;
  int g = (b << 4) + a;
  float w = ww[g * 16 + slot];
  const float* wrow = wp + (long)g * 448;
  float er = 1.0f / (1.0f + __expf(-wrow[129 + m]));
  float ad = tanhf(wrow[257 + m]);
  float mv = mem[idx];
  mem[idx] = mv * (1.0f - w * er) + w * ad;
}

// ---------------------------------------------------------------------------
// Read-head addressing + weighted read, fused. One block (128 thr) per (h,g).
// Stages the whole 128 KB scene memory tile in LDS (WGP has 320 KB), computes
// per-slot cosine sims + norms from LDS, block softmax over 256 slots with the
// agent's own band masked out, then reads[h,g,:] = sum_n w[n]*mem[n,:].
// Dynamic LDS = 256*128 (mem) + 128 (key) + 256 (weights) floats.
// rp row layout: 5 heads of [k(128)|beta], ld=704.
// ---------------------------------------------------------------------------
extern __shared__ float sm[];
__global__ void read_kernel(const float* __restrict__ rp, const float* __restrict__ mem,
                            float* __restrict__ reads, float* __restrict__ rbuf) {
  __shared__ float red[128];
  int blk = blockIdx.x;
  int h = blk >> 10;
  int g = blk & 1023;
  int b = g >> 4, a = g & 15;
  int t = threadIdx.x;
  float* smem = sm;
  float* kr = sm + 32768;
  float* wv = sm + 32768 + 128;
  const float* mb = mem + (long)b * 256 * 128;
  for (int i = t; i < 256 * 128; i += 128) smem[i] = mb[i];
  const float* rrow = rp + (long)g * 704 + h * 129;
  float kv = tanhf(rrow[t]);
  kr[t] = kv;
  red[t] = kv * kv;
  __syncthreads();
  for (int s = 64; s > 0; s >>= 1) { if (t < s) red[t] += red[t + s]; __syncthreads(); }
  float knorm = sqrtf(red[0]) + 1e-16f;
  float bet = rrow[128];
  float beta = (bet > 20.0f) ? bet : log1pf(__expf(bet));
#pragma unroll
  for (int sidx = 0; sidx < 2; ++sidx) {
    int n = t + sidx * 128;
    float logit = -3.0e38f;  // own band masked out (mask_r)
    if ((n >> 4) != a) {
      const float* mrow = smem + n * 128;
      float dot = 0.0f, nn = 0.0f;
      for (int m = 0; m < 128; ++m) { float mv = mrow[m]; dot += kr[m] * mv; nn += mv * mv; }
      logit = beta * dot / (knorm * (sqrtf(nn) + 1e-16f));
    }
    wv[n] = logit;
  }
  __syncthreads();
  float lmx = fmaxf(wv[t], wv[t + 128]);
  red[t] = lmx; __syncthreads();
  for (int s = 64; s > 0; s >>= 1) { if (t < s) red[t] = fmaxf(red[t], red[t + s]); __syncthreads(); }
  float mx = red[0]; __syncthreads();
  float e0 = __expf(wv[t] - mx), e1 = __expf(wv[t + 128] - mx);
  red[t] = e0 + e1; __syncthreads();
  for (int s = 64; s > 0; s >>= 1) { if (t < s) red[t] += red[t + s]; __syncthreads(); }
  float inv = 1.0f / red[0];
  __syncthreads();
  wv[t] = e0 * inv;
  wv[t + 128] = e1 * inv;
  __syncthreads();
  float acc = 0.0f;
  for (int n = 0; n < 256; ++n) acc += wv[n] * smem[n * 128 + t];
  reads[((long)h * 1024 + g) * 128 + t] = acc;
  if (h == 0) rbuf[(long)g * 128 + t] = acc;  // r = reads[0]
}

// dec_in[(h*1024+g)] = [h_enc(g) | reads(h,g)]
__global__ void decin_kernel(const float* __restrict__ henc, const float* __restrict__ reads,
                             float* __restrict__ dec_in) {
  long idx = (long)blockIdx.x * blockDim.x + threadIdx.x;
  if (idx >= (long)5120 * 384) return;
  int row = (int)(idx / 384), j = (int)(idx % 384);
  int h = row >> 10, g = row & 1023;
  dec_in[idx] = (j < 256) ? henc[(long)g * 256 + j] : reads[((long)h * 1024 + g) * 128 + (j - 256)];
}

// out = hf @ out_w^T + out_b   (N=2; plain VALU dot, WMMA would be 87% padding)
__global__ void out_kernel(const float* __restrict__ hf, const float* __restrict__ out_w,
                           const float* __restrict__ out_b, float* __restrict__ outs) {
  int row = blockIdx.x * blockDim.x + threadIdx.x;
  if (row >= 5120) return;
  const float* hr = hf + (long)row * 384;
  float a0 = out_b[0], a1 = out_b[1];
  for (int k = 0; k < 384; ++k) {
    float hv = hr[k];
    a0 += out_w[k] * hv;
    a1 += out_w[384 + k] * hv;
  }
  outs[(long)row * 2 + 0] = a0;
  outs[(long)row * 2 + 1] = a1;
}

// pred = past[:, -1] + cumsum(pred_rel); d_out = [pred | pred_rel] both (A,H,30,2)
__global__ void final_kernel(const float* __restrict__ outs, const float* __restrict__ past,
                             float* __restrict__ out) {
  int idx = blockIdx.x * blockDim.x + threadIdx.x;
  if (idx >= 1024 * 5 * 2) return;
  int c = idx & 1;
  int h = (idx >> 1) % 5;
  int g = idx / 10;
  float accum = past[((long)g * 20 + 19) * 2 + c];
  float* pred = out;
  float* prel = out + (long)1024 * 5 * 30 * 2;
  for (int t = 0; t < 30; ++t) {
    float v = outs[(((long)t * 5 + h) * 1024 + g) * 2 + c];
    accum += v;
    long o = (((long)g * 5 + h) * 30 + t) * 2 + c;
    pred[o] = accum;
    prel[o] = v;
  }
}

// ---------------------------------------------------------------------------
// host launcher
// ---------------------------------------------------------------------------
static inline int nblk(long n) { return (int)((n + 255) / 256); }

extern "C" void kernel_launch(void* const* d_in, const int* in_sizes, int n_in,
                              void* d_out, int out_size, void* d_ws, size_t ws_size,
                              hipStream_t stream) {
  (void)in_sizes; (void)n_in; (void)out_size; (void)ws_size;
  const float* past     = (const float*)d_in[0];
  const float* past_rel = (const float*)d_in[1];
  const float* ep_w1 = (const float*)d_in[2];
  const float* ep_b1 = (const float*)d_in[3];
  const float* ep_w2 = (const float*)d_in[4];
  const float* ep_b2 = (const float*)d_in[5];
  const float* er_w1 = (const float*)d_in[6];
  const float* er_b1 = (const float*)d_in[7];
  const float* er_w2 = (const float*)d_in[8];
  const float* er_b2 = (const float*)d_in[9];
  const float* enc_wih = (const float*)d_in[10];
  const float* enc_whh = (const float*)d_in[11];
  const float* enc_bih = (const float*)d_in[12];
  const float* enc_bhh = (const float*)d_in[13];
  const float* rel_wih = (const float*)d_in[14];
  const float* rel_whh = (const float*)d_in[15];
  const float* rel_bih = (const float*)d_in[16];
  const float* rel_bhh = (const float*)d_in[17];
  const float* dec_wih = (const float*)d_in[18];
  const float* dec_whh = (const float*)d_in[19];
  const float* dec_bih = (const float*)d_in[20];
  const float* dec_bhh = (const float*)d_in[21];
  const float* out_w = (const float*)d_in[22];
  const float* out_b = (const float*)d_in[23];
  const float* read_w = (const float*)d_in[24];
  const float* read_b = (const float*)d_in[25];
  const float* write_w = (const float*)d_in[26];
  const float* write_b = (const float*)d_in[27];
  const float* init_r = (const float*)d_in[28];
  const float* h_bias = (const float*)d_in[29];
  const float* h_rel_bias = (const float*)d_in[30];
  const float* h_fut_bias = (const float*)d_in[31];
  const float* mem_bias = (const float*)d_in[32];

  // ---- workspace carve-up (all fits in L2: ~100 MB) ----
  char* ws = (char*)d_ws;
  size_t off = 0;
  auto alloc = [&](size_t bytes) -> void* {
    void* p = ws + off;
    off = (off + bytes + 255) & ~(size_t)255;
    return p;
  };
  float* embP   = (float*)alloc((size_t)20 * 1024 * 16 * 4);
  float* embR   = (float*)alloc((size_t)20 * 1024 * 16 * 4);
  float* zeroE  = (float*)alloc((size_t)1024 * 16 * 4);
  float* xcat   = (float*)alloc((size_t)1024 * 144 * 4);
  float* giE    = (float*)alloc((size_t)1024 * 768 * 4);
  float* ghE    = (float*)alloc((size_t)1024 * 768 * 4);
  float* giR    = (float*)alloc((size_t)1024 * 768 * 4);
  float* ghR    = (float*)alloc((size_t)1024 * 768 * 4);
  float* hS     = (float*)alloc((size_t)1024 * 256 * 4);
  float* hrS    = (float*)alloc((size_t)1024 * 256 * 4);
  float* hfS    = (float*)alloc((size_t)5120 * 384 * 4);
  float* rS     = (float*)alloc((size_t)1024 * 128 * 4);
  float* memS   = (float*)alloc((size_t)64 * 256 * 128 * 4);
  float* wpB    = (float*)alloc((size_t)1024 * 448 * 4);
  float* wwB    = (float*)alloc((size_t)1024 * 16 * 4);
  float* rpB    = (float*)alloc((size_t)1024 * 704 * 4);
  float* readsB = (float*)alloc((size_t)5 * 1024 * 128 * 4);
  float* decIn  = (float*)alloc((size_t)5120 * 384 * 4);
  float* giD    = (float*)alloc((size_t)5120 * 1152 * 4);
  float* ghD    = (float*)alloc((size_t)5120 * 1152 * 4);
  float* outsB  = (float*)alloc((size_t)30 * 5120 * 2 * 4);
  __bf16* enc_wih_p = (__bf16*)alloc((size_t)5 * 48 * 512 * 2);
  __bf16* enc_whh_p = (__bf16*)alloc((size_t)8 * 48 * 512 * 2);
  __bf16* rel_wih_p = (__bf16*)alloc((size_t)1 * 48 * 512 * 2);
  __bf16* rel_whh_p = (__bf16*)alloc((size_t)8 * 48 * 512 * 2);
  __bf16* dec_wih_p = (__bf16*)alloc((size_t)12 * 72 * 512 * 2);
  __bf16* dec_whh_p = (__bf16*)alloc((size_t)12 * 72 * 512 * 2);
  __bf16* write_w_p = (__bf16*)alloc((size_t)8 * 28 * 512 * 2);
  __bf16* read_w_p  = (__bf16*)alloc((size_t)8 * 44 * 512 * 2);

  auto pack = [&](const float* W, __bf16* P, int K, int Nout, int Kt, int Nt) {
    long tot = (long)Kt * Nt * 32;
    pack_b_kernel<<<nblk(tot), 256, 0, stream>>>(W, P, K, Nout, Kt, Nt);
  };
  auto gemm = [&](const float* Amat, int lda, int Mrows, const __bf16* Bp, int K, int Kt,
                  int Ntiles, const float* bias, int Nreal, float* Cmat, int ldc) {
    long waves = (long)(Mrows / 16) * (Ntiles / 4);
    wmma_gemm_bias<<<nblk(waves * 32), 256, 0, stream>>>(Amat, lda, Mrows, Bp, K, Kt, Ntiles,
                                                         bias, Nreal, Cmat, ldc);
  };

  // ---- one-time per call: weight packing, embeddings, state init ----
  pack(enc_wih, enc_wih_p, 144, 768, 5, 48);
  pack(enc_whh, enc_whh_p, 256, 768, 8, 48);
  pack(rel_wih, rel_wih_p, 16, 768, 1, 48);
  pack(rel_whh, rel_whh_p, 256, 768, 8, 48);
  pack(dec_wih, dec_wih_p, 384, 1152, 12, 72);
  pack(dec_whh, dec_whh_p, 384, 1152, 12, 72);
  pack(write_w, write_w_p, 256, 385, 8, 28);
  pack(read_w, read_w_p, 256, 645, 8, 44);

  zero_kernel<<<nblk(1024 * 16), 256, 0, stream>>>(zeroE, 1024 * 16);
  bcast_kernel<<<nblk((long)1024 * 256), 256, 0, stream>>>(hS, h_bias, 1024, 256);
  bcast_kernel<<<nblk((long)1024 * 256), 256, 0, stream>>>(hrS, h_rel_bias, 1024, 256);
  bcast_kernel<<<nblk((long)5120 * 384), 256, 0, stream>>>(hfS, h_fut_bias, 5120, 384);
  bcast_kernel<<<nblk((long)1024 * 128), 256, 0, stream>>>(rS, init_r, 1024, 128);
  bcast_kernel<<<nblk((long)64 * 256 * 128), 256, 0, stream>>>(memS, mem_bias, 64, 256 * 128);

  embed_kernel<<<nblk(1024 * 20), 256, 0, stream>>>(past, ep_w1, ep_b1, ep_w2, ep_b2, embP);
  embed_kernel<<<nblk(1024 * 20), 256, 0, stream>>>(past_rel, er_w1, er_b1, er_w2, er_b2, embR);

  const size_t read_lds = (size_t)(256 * 128 + 128 + 256) * sizeof(float);

  // ---- 50 recurrent steps (20 past + 30 future) ----
  for (int ti = 0; ti < 50; ++ti) {
    bool fut = ti >= 20;
    const float* embPt = fut ? zeroE : (embP + (size_t)ti * 1024 * 16);
    const float* embRt = fut ? zeroE : (embR + (size_t)ti * 1024 * 16);

    // encoder GRU: x = [emb | r]
    xcat_kernel<<<nblk(1024 * 144), 256, 0, stream>>>(embPt, rS, xcat);
    gemm(xcat, 144, 1024, enc_wih_p, 144, 5, 48, enc_bih, 768, giE, 768);
    gemm(hS, 256, 1024, enc_whh_p, 256, 8, 48, enc_bhh, 768, ghE, 768);
    gru_combine_kernel<<<nblk((long)1024 * 256), 256, 0, stream>>>(giE, ghE, hS, 1024, 256);

    // relative GRU
    gemm(embRt, 16, 1024, rel_wih_p, 16, 1, 48, rel_bih, 768, giR, 768);
    gemm(hrS, 256, 1024, rel_whh_p, 256, 8, 48, rel_bhh, 768, ghR, 768);
    gru_combine_kernel<<<nblk((long)1024 * 256), 256, 0, stream>>>(giR, ghR, hrS, 1024, 256);

    // write head + memory update
    gemm(hS, 256, 1024, write_w_p, 256, 8, 28, write_b, 385, wpB, 448);
    write_addr_kernel<<<1024, 128, 0, stream>>>(wpB, memS, wwB);
    mem_update_kernel<<<nblk((long)64 * 256 * 128), 256, 0, stream>>>(memS, wwB, wpB);

    // read heads (address the UPDATED memory) + fused weighted read
    gemm(hrS, 256, 1024, read_w_p, 256, 8, 44, read_b, 645, rpB, 704);
    read_kernel<<<5120, 128, read_lds, stream>>>(rpB, memS, readsB, rS);

    // per-mode decoder GRU
    decin_kernel<<<nblk((long)5120 * 384), 256, 0, stream>>>(hS, readsB, decIn);
    gemm(decIn, 384, 5120, dec_wih_p, 384, 12, 72, dec_bih, 1152, giD, 1152);
    gemm(hfS, 384, 5120, dec_whh_p, 384, 12, 72, dec_bhh, 1152, ghD, 1152);
    gru_combine_kernel<<<nblk((long)5120 * 384), 256, 0, stream>>>(giD, ghD, hfS, 5120, 384);

    if (fut)
      out_kernel<<<nblk(5120), 256, 0, stream>>>(hfS, out_w, out_b,
                                                 outsB + (size_t)(ti - 20) * 5120 * 2);
  }

  final_kernel<<<nblk(1024 * 5 * 2), 256, 0, stream>>>(outsB, past, (float*)d_out);
}
